// FastAttention_10746008174814
// MI455X (gfx1250) — matrix operand order repaired and checked by reading, not compiled, not run on hardware
//
#include <hip/hip_runtime.h>
#include <math.h>

// ---------------------------------------------------------------------------
// Performer (FAVOR+) fast attention for MI455X / gfx1250.
// All GEMMs on v_wmma_f32_16x16x32_f16 (f32 accumulate); exp/max/diag/1/D in f32.
// wave32, 256-thread blocks (8 waves), WMMA fragments hand-built from LDS per
// the CDNA5 ISA layouts (A 16-bit 16x32, B 32x16, C/D 16x16 f32).
// ---------------------------------------------------------------------------

typedef __attribute__((ext_vector_type(16))) _Float16 v16h;
typedef __attribute__((ext_vector_type(8)))  _Float16 v8h;
typedef __attribute__((ext_vector_type(8)))  float    v8f;

namespace {
constexpr int BB  = 4, HH = 8, NN = 8192, DD = 64, MM = 266;
constexpr int BH  = BB * HH;
constexpr int MP  = 288;            // M padded to 18 tiles of 16 (multiple of 32)
constexpr int MT  = MP / 16;        // 18 m-tiles
constexpr int NCH = 128;            // n rows per chunk (8 waves x 16 rows)
constexpr int SPLIT = 8;            // n-segments per (b,h) in context kernel
constexpr float DN    = 0.35355339059327373f;  // 64^-0.25
constexpr float RATIO = 0.06131393f;           // 266^-0.5
constexpr float EPSF  = 1e-4f;

// workspace layout (floats)
constexpr int WS_GMAX = 0;                  // [16] (only [0] used)
constexpr int WS_KSUM = 16;                 // [BH][MP]
constexpr int WS_CTX  = WS_KSUM + BH * MP;  // [BH][MP][DD]
constexpr int WS_TOT  = WS_CTX + BH * MP * DD;

constexpr int LDT = NCH + 8;  // 136 halves: leading dim of [*][n] transposed tiles
constexpr int LDM = MP + 8;   // 296 halves: leading dim of [*][m] tiles
} // namespace

// ---- WMMA fragment builders (LDS, 16B-aligned v8h loads) -------------------
// A (16-bit 16x32): lane(l) elem(e) -> K = (e<8?0:16) + 8*(l>>4) + (e&7)
__device__ __forceinline__ v16h frag_a(const _Float16* rowp, int k0, int h) {
  v8h lo = *(const v8h*)(rowp + k0 + 8 * h);
  v8h hi = *(const v8h*)(rowp + k0 + 16 + 8 * h);
  return __builtin_shufflevector(lo, hi, 0, 1, 2, 3, 4, 5, 6, 7,
                                 8, 9, 10, 11, 12, 13, 14, 15);
}
// B (32x16): lane(l) elem(e) -> K = (e<8?0:16) + 8*(l>>4) + (e&7)
__device__ __forceinline__ v16h frag_b(const _Float16* colp, int k0, int h) {
  v8h lo = *(const v8h*)(colp + k0 + 8 * h);
  v8h hi = *(const v8h*)(colp + k0 + 16 + 8 * h);
  return __builtin_shufflevector(lo, hi, 0, 1, 2, 3, 4, 5, 6, 7,
                                 8, 9, 10, 11, 12, 13, 14, 15);
}
__device__ __forceinline__ v8f wmma_f16(v16h a, v16h b, v8f c) {
  return __builtin_amdgcn_wmma_f32_16x16x32_f16(false, a, false, b,
                                                (short)0, c, false, false);
}

__device__ __forceinline__ void atomicMaxF(float* addr, float val) {
  if (val >= 0.0f) atomicMax((int*)addr, __float_as_int(val));
  else             atomicMin((unsigned int*)addr, __float_as_uint(val));
}

// ---------------------------------------------------------------------------
__global__ void k_init(float* __restrict__ ws) {
  int i = blockIdx.x * 256 + threadIdx.x;
  if (i < WS_TOT) ws[i] = (i == 0) ? -__builtin_inff() : 0.0f;
}

// ---------------------------------------------------------------------------
// Kernel 1: global max of k_dash = (dn*k) @ proj^T  (raw, pre-diag, per ref)
__global__ __launch_bounds__(256) void k_gmax(const float* __restrict__ kk,
                                              const float* __restrict__ proj,
                                              float* __restrict__ ws) {
  extern __shared__ __align__(16) char smem[];
  _Float16* projL = (_Float16*)(smem);            // [MP][64]
  _Float16* kL    = (_Float16*)(smem + 36864);    // [NCH][64]
  float*    red   = (float*)(smem + 53248);       // [8]

  const int tid = threadIdx.x;
  const int bh  = blockIdx.x / (NN / NCH);
  const int n0  = (blockIdx.x % (NN / NCH)) * NCH;
  const float* kbh = kk + (size_t)bh * NN * DD;

  for (int i = tid; i < MP * DD; i += 256) {
    int m = i >> 6, d = i & 63;
    projL[i] = (_Float16)((m < MM) ? proj[m * DD + d] : 0.0f);
  }
  for (int i = tid; i < NCH * DD; i += 256)
    kL[i] = (_Float16)(DN * kbh[(size_t)n0 * DD + i]);
  __syncthreads();

  const int lane = tid & 31, w = tid >> 5, lid = lane & 15, h = lane >> 4;
  const _Float16* arow = kL + (w * 16 + lid) * DD;
  float mymax = -__builtin_inff();
  for (int mt = 0; mt < MT; ++mt) {
    v8f acc = {};
    const _Float16* bcol = projL + (mt * 16 + lid) * DD;
    acc = wmma_f16(frag_a(arow, 0, h),  frag_b(bcol, 0, h),  acc);
    acc = wmma_f16(frag_a(arow, 32, h), frag_b(bcol, 32, h), acc);
    const bool valid = (mt * 16 + lid) < MM;
#pragma unroll
    for (int r = 0; r < 8; ++r)
      mymax = valid ? fmaxf(mymax, acc[r]) : mymax;
  }
#pragma unroll
  for (int off = 16; off > 0; off >>= 1)
    mymax = fmaxf(mymax, __shfl_xor(mymax, off, 32));
  if (lane == 0) red[w] = mymax;
  __syncthreads();
  if (tid == 0) {
    float bm = red[0];
    for (int i = 1; i < 8; ++i) bm = fmaxf(bm, red[i]);
    atomicMaxF(ws + WS_GMAX, bm);
  }
}

// ---------------------------------------------------------------------------
// Kernel 2: kp = ratio*(exp(k_dash - diag - gmax)+eps); context += kp^T @ v;
//           ksum += sum_n kp.   One block per (b,h, n-segment of 1024 rows).
__global__ __launch_bounds__(256) void k_context(const float* __restrict__ kk,
                                                 const float* __restrict__ vv,
                                                 const float* __restrict__ proj,
                                                 float* __restrict__ ws) {
  extern __shared__ __align__(16) char smem[];
  _Float16* projL = (_Float16*)(smem);             // [MP][64]     36864 B
  _Float16* kL    = (_Float16*)(smem + 36864);     // [NCH][64]    16384 B
  _Float16* vT    = (_Float16*)(smem + 53248);     // [64][LDT]    17408 B
  _Float16* kpT   = (_Float16*)(smem + 70656);     // [MP][LDT]    78336 B
  float*    sq    = (float*)(smem + 148992);       // [256]
  float*    diag  = (float*)(smem + 150016);       // [128]

  const int tid = threadIdx.x;
  const int bh  = blockIdx.x / SPLIT;
  const int seg = blockIdx.x % SPLIT;
  const float* kbh = kk + (size_t)bh * NN * DD;
  const float* vbh = vv + (size_t)bh * NN * DD;
  const float gmax = ws[WS_GMAX];

  for (int i = tid; i < MP * DD; i += 256) {
    int m = i >> 6, d = i & 63;
    projL[i] = (_Float16)((m < MM) ? proj[m * DD + d] : 0.0f);
  }

  const int lane = tid & 31, w = tid >> 5, lid = lane & 15, h = lane >> 4;

  v8f accT[9];
#pragma unroll
  for (int i = 0; i < 9; ++i) accT[i] = (v8f){};
  float ks0 = 0.0f, ks1 = 0.0f;

  for (int c = 0; c < (NN / SPLIT) / NCH; ++c) {
    const int n0 = seg * (NN / SPLIT) + c * NCH;
    __syncthreads();  // protect LDS tiles vs previous-iteration readers

    {  // stage dn*k (f16) + row sum-of-squares
      const int row = tid >> 1, c0 = (tid & 1) * 32;
      const float* src = kbh + (size_t)(n0 + row) * DD + c0;
      float s = 0.0f;
#pragma unroll
      for (int j = 0; j < 32; j += 4) {
        float4 x = *(const float4*)(src + j);
        s += x.x * x.x + x.y * x.y + x.z * x.z + x.w * x.w;
        _Float16* dst = kL + row * DD + c0 + j;
        dst[0] = (_Float16)(DN * x.x); dst[1] = (_Float16)(DN * x.y);
        dst[2] = (_Float16)(DN * x.z); dst[3] = (_Float16)(DN * x.w);
      }
      sq[tid] = s;
    }
    for (int i = tid; i < NCH * DD; i += 256) {  // stage v transposed
      int row = i >> 6, d = i & 63;
      vT[d * LDT + row] = (_Float16)vbh[(size_t)(n0 + row) * DD + d];
    }
    __syncthreads();
    if (tid < 128) diag[tid] = 0.5f * DN * DN * (sq[2 * tid] + sq[2 * tid + 1]);
    __syncthreads();

    // stage 1: kp for this wave's 16 rows, write transposed [m][n]
    const _Float16* arow = kL + (w * 16 + lid) * DD;
    for (int mt = 0; mt < MT; ++mt) {
      v8f acc = {};
      const _Float16* bcol = projL + (mt * 16 + lid) * DD;
      acc = wmma_f16(frag_a(arow, 0, h),  frag_b(bcol, 0, h),  acc);
      acc = wmma_f16(frag_a(arow, 32, h), frag_b(bcol, 32, h), acc);
      const int m = mt * 16 + lid;
      v8h outv = {};
#pragma unroll
      for (int r = 0; r < 8; ++r) {
        const int row = w * 16 + r + 8 * h;
        float e  = __expf(acc[r] - diag[row] - gmax);
        float kp = (m < MM) ? RATIO * (e + EPSF) : 0.0f;
        outv[r] = (_Float16)kp;
      }
      *(v8h*)(kpT + m * LDT + w * 16 + 8 * h) = outv;  // 16B aligned
    }
    __syncthreads();

    {  // ksum partials: thread t sums column m=t (and m=256+t for t<32)
      const _Float16* col = kpT + tid * LDT;
      float s = 0.0f;
      for (int n = 0; n < NCH; n += 8) {
        v8h x = *(const v8h*)(col + n);
#pragma unroll
        for (int j = 0; j < 8; ++j) s += (float)x[j];
      }
      ks0 += s;
      if (tid < MP - 256) {
        const _Float16* col2 = kpT + (256 + tid) * LDT;
        float s2 = 0.0f;
        for (int n = 0; n < NCH; n += 8) {
          v8h x = *(const v8h*)(col2 + n);
#pragma unroll
          for (int j = 0; j < 8; ++j) s2 += (float)x[j];
        }
        ks1 += s2;
      }
    }

    // stage 2: context[m,e] += kp^T @ v over this chunk; 9 tiles per wave
#pragma unroll
    for (int i = 0; i < 9; ++i) {
      const int t = w + 8 * i;          // 0..71 == 18 mtiles x 4 etiles
      const int mt = t >> 2, et = t & 3;
      const _Float16* ab = kpT + (mt * 16 + lid) * LDT;
      const _Float16* bb = vT + (et * 16 + lid) * LDT;
      v8f a = accT[i];
#pragma unroll
      for (int k0 = 0; k0 < NCH; k0 += 32)
        a = wmma_f16(frag_a(ab, k0, h), frag_b(bb, k0, h), a);
      accT[i] = a;
    }
  }

  float* ksum = ws + WS_KSUM + bh * MP;
  atomicAdd(ksum + tid, ks0);
  if (tid < MP - 256) atomicAdd(ksum + 256 + tid, ks1);
  float* ctx = ws + WS_CTX + (size_t)bh * MP * DD;
#pragma unroll
  for (int i = 0; i < 9; ++i) {
    const int t = w + 8 * i;
    const int mt = t >> 2, et = t & 3;
#pragma unroll
    for (int r = 0; r < 8; ++r) {
      const int mrow = mt * 16 + r + 8 * h;
      atomicAdd(ctx + mrow * DD + et * 16 + lid, accT[i][r]);
    }
  }
}

// ---------------------------------------------------------------------------
// Kernel 3: qp (per-row max) and out = (qp @ context) / (qp . ksum)
__global__ __launch_bounds__(256) void k_out(const float* __restrict__ qq,
                                             const float* __restrict__ proj,
                                             const float* __restrict__ ws,
                                             float* __restrict__ out) {
  extern __shared__ __align__(16) char smem[];
  _Float16* projL = (_Float16*)(smem);             // [MP][64]    36864 B
  _Float16* qL    = (_Float16*)(smem + 36864);     // [NCH][64]   16384 B
  _Float16* ctxT  = (_Float16*)(smem + 53248);     // [64][LDM]   37888 B
  _Float16* qpL   = (_Float16*)(smem + 91136);     // [NCH][LDM]  75776 B
  float*    ksumL = (float*)(smem + 166912);       // [MP]
  float*    sq    = (float*)(smem + 168064);       // [256]
  float*    diag  = (float*)(smem + 169088);       // [128]

  const int tid = threadIdx.x;
  const int bh  = blockIdx.x / (NN / NCH);
  const int n0  = (blockIdx.x % (NN / NCH)) * NCH;
  const float* qbh  = qq + (size_t)bh * NN * DD;
  const float* ctx  = ws + WS_CTX + (size_t)bh * MP * DD;
  const float* ksum = ws + WS_KSUM + bh * MP;

  for (int i = tid; i < MP * DD; i += 256) {
    int m = i >> 6, d = i & 63;
    projL[i] = (_Float16)((m < MM) ? proj[m * DD + d] : 0.0f);
  }
  for (int i = tid; i < MP * DD; i += 256) {  // context transposed [e][m]
    int m = i >> 6, e = i & 63;
    ctxT[e * LDM + m] = (_Float16)ctx[i];
  }
  for (int i = tid; i < MP; i += 256) ksumL[i] = ksum[i];
  {  // stage dn*q + row sum-of-squares
    const int row = tid >> 1, c0 = (tid & 1) * 32;
    const float* src = qbh + (size_t)(n0 + row) * DD + c0;
    float s = 0.0f;
#pragma unroll
    for (int j = 0; j < 32; j += 4) {
      float4 x = *(const float4*)(src + j);
      s += x.x * x.x + x.y * x.y + x.z * x.z + x.w * x.w;
      _Float16* dst = qL + row * DD + c0 + j;
      dst[0] = (_Float16)(DN * x.x); dst[1] = (_Float16)(DN * x.y);
      dst[2] = (_Float16)(DN * x.z); dst[3] = (_Float16)(DN * x.w);
    }
    sq[tid] = s;
  }
  __syncthreads();
  if (tid < 128) diag[tid] = 0.5f * DN * DN * (sq[2 * tid] + sq[2 * tid + 1]);
  __syncthreads();

  const int lane = tid & 31, w = tid >> 5, lid = lane & 15, h = lane >> 4;
  const _Float16* arow = qL + (w * 16 + lid) * DD;

  // pass 1: per-row max over valid m
  float rmax[8];
#pragma unroll
  for (int r = 0; r < 8; ++r) rmax[r] = -__builtin_inff();
  for (int mt = 0; mt < MT; ++mt) {
    v8f acc = {};
    const _Float16* bcol = projL + (mt * 16 + lid) * DD;
    acc = wmma_f16(frag_a(arow, 0, h),  frag_b(bcol, 0, h),  acc);
    acc = wmma_f16(frag_a(arow, 32, h), frag_b(bcol, 32, h), acc);
    const bool valid = (mt * 16 + lid) < MM;
#pragma unroll
    for (int r = 0; r < 8; ++r)
      rmax[r] = valid ? fmaxf(rmax[r], acc[r]) : rmax[r];
  }
#pragma unroll
  for (int off = 1; off < 16; off <<= 1)
#pragma unroll
    for (int r = 0; r < 8; ++r)
      rmax[r] = fmaxf(rmax[r], __shfl_xor(rmax[r], off, 32));

  // pass 2: qp + D = sum_m qp*ksum
  float dpart[8];
#pragma unroll
  for (int r = 0; r < 8; ++r) dpart[r] = 0.0f;
  for (int mt = 0; mt < MT; ++mt) {
    v8f acc = {};
    const _Float16* bcol = projL + (mt * 16 + lid) * DD;
    acc = wmma_f16(frag_a(arow, 0, h),  frag_b(bcol, 0, h),  acc);
    acc = wmma_f16(frag_a(arow, 32, h), frag_b(bcol, 32, h), acc);
    const int m = mt * 16 + lid;
    const float ksm = ksumL[m];
#pragma unroll
    for (int r = 0; r < 8; ++r) {
      const int row = w * 16 + r + 8 * h;
      float e  = __expf(acc[r] - diag[row] - rmax[r]);
      float qp = (m < MM) ? RATIO * (e + EPSF) : 0.0f;
      qpL[row * LDM + m] = (_Float16)qp;
      dpart[r] += qp * ksm;
    }
  }
#pragma unroll
  for (int off = 1; off < 16; off <<= 1)
#pragma unroll
    for (int r = 0; r < 8; ++r) dpart[r] += __shfl_xor(dpart[r], off, 32);
  float dinv[8];
#pragma unroll
  for (int r = 0; r < 8; ++r) dinv[r] = 1.0f / dpart[r];
  __syncthreads();

  // stage 2: out = dinv * (qp @ context), K = MP = 288
  float* obh = out + (size_t)bh * NN * DD;
  const _Float16* aq = qpL + (w * 16 + lid) * LDM;
#pragma unroll
  for (int et = 0; et < 4; ++et) {
    v8f acc = {};
    const _Float16* bb = ctxT + (et * 16 + lid) * LDM;
#pragma unroll
    for (int k0 = 0; k0 < MP; k0 += 32)
      acc = wmma_f16(frag_a(aq, k0, h), frag_b(bb, k0, h), acc);
#pragma unroll
    for (int r = 0; r < 8; ++r) {
      const int row = w * 16 + r + 8 * h;
      obh[(size_t)(n0 + row) * DD + et * 16 + lid] = dinv[r] * acc[r];
    }
  }
}

// ---------------------------------------------------------------------------
extern "C" void kernel_launch(void* const* d_in, const int* in_sizes, int n_in,
                              void* d_out, int out_size, void* d_ws, size_t ws_size,
                              hipStream_t stream) {
  (void)in_sizes; (void)n_in; (void)out_size; (void)ws_size;
  const float* q    = (const float*)d_in[0];
  const float* k    = (const float*)d_in[1];
  const float* v    = (const float*)d_in[2];
  const float* proj = (const float*)d_in[3];
  float* out = (float*)d_out;
  float* ws  = (float*)d_ws;

  constexpr size_t SMEM1 = 53312;    // proj + k-chunk + reduce
  constexpr size_t SMEM2 = 150528;   // proj + k + vT + kpT + stats
  constexpr size_t SMEM3 = 169600;   // proj + q + ctxT + qpL + ksum + stats

  k_init<<<(WS_TOT + 255) / 256, 256, 0, stream>>>(ws);
  k_gmax<<<BH * (NN / NCH), 256, SMEM1, stream>>>(k, proj, ws);
  k_context<<<BH * SPLIT, 256, SMEM2, stream>>>(k, v, proj, ws);
  k_out<<<BH * (NN / NCH), 256, SMEM3, stream>>>(q, proj, ws, out);
}
